// MGKAttention_5738076307878
// MI455X (gfx1250) — matrix operand order repaired and checked by reading, not compiled, not run on hardware
//
#include <hip/hip_runtime.h>

typedef __attribute__((ext_vector_type(16))) _Float16 v16h;
typedef __attribute__((ext_vector_type(8)))  _Float16 v8h;
typedef __attribute__((ext_vector_type(4)))  _Float16 v4h;
typedef __attribute__((ext_vector_type(8)))  float    v8f;

#define TT 1024
#define BB 4
#define EE 1024
#define HH 16
#define HD 64
#define TB 4096          // T*B rows
#define RPA 65536        // rows per head-major array: 64*1024

static __device__ __forceinline__ v16h cat8(v8h lo, v8h hi) {
  return __builtin_shufflevector(lo, hi, 0,1,2,3,4,5,6,7,8,9,10,11,12,13,14,15);
}

// A-matrix fragment 16x32 f16: lanes 0-15 hold M=0..15 with K = [0..7],[16..23];
// lanes 16-31 same rows, K = [8..15],[24..31].
static __device__ __forceinline__ v16h frag_a(const _Float16* p, int ld, int k0, int lane) {
  const int r    = lane & 15;
  const int koff = (lane >> 4) << 3;
  const _Float16* q = p + r * ld + k0 + koff;
  return cat8(*(const v8h*)q, *(const v8h*)(q + 16));
}

// B-matrix fragment 32x16 f16 from a source stored transposed (BT[n][k], k contiguous):
// lanes 0-15: column n, K = [0..7],[16..23]; lanes 16-31: column n, K = [8..15],[24..31].
static __device__ __forceinline__ v16h frag_b(const _Float16* p, int ld, int k0, int lane) {
  const int n  = lane & 15;
  const int kh = (lane >> 4) << 3;
  const _Float16* q = p + n * ld + k0 + kh;
  return cat8(*(const v8h*)q, *(const v8h*)(q + 16));
}

static __device__ __forceinline__ v8f wmma_f16(v16h a, v16h b, v8f c) {
  return __builtin_amdgcn_wmma_f32_16x16x32_f16(false, a, false, b, (short)0, c, false, false);
}

// ---- CDNA5 async global->LDS helpers (ASYNCcnt-tracked, overlap with compute)
static __device__ __forceinline__ unsigned lds_addr_of(const void* p) {
  // generic pointers to LDS carry the DS byte offset in their low 32 bits
  return (unsigned)(uintptr_t)p;
}
static __device__ __forceinline__ void async_copy_b128(unsigned lds, unsigned voff,
                                                       const void* sbase) {
  asm volatile("global_load_async_to_lds_b128 %0, %1, %2"
               :: "v"(lds), "v"(voff), "s"(sbase) : "memory");
}
static __device__ __forceinline__ void wait_async0() {
  asm volatile("s_wait_asynccnt 0x0" ::: "memory");
}
static __device__ __forceinline__ void wait_ds0() {
  asm volatile("s_wait_dscnt 0x0" ::: "memory");
}

// ---------------------------------------------------------------------------
// Kernel 1: projections Y = X @ W^T + bias, output in head-major f16 layout
// grid: (N/128=8, M/128=32, 4)   block: 256
// ---------------------------------------------------------------------------
__global__ __launch_bounds__(256) void proj_kernel(
    const float* __restrict__ query, const float* __restrict__ key,
    const float* __restrict__ value,
    const float* __restrict__ Wq, const float* __restrict__ Wk1,
    const float* __restrict__ Wk2, const float* __restrict__ Wv,
    const float* __restrict__ bias,
    _Float16* __restrict__ qh, _Float16* __restrict__ k1h,
    _Float16* __restrict__ k2h, _Float16* __restrict__ vh)
{
  const int which = blockIdx.z;
  const float* X; const float* W; _Float16* dst;
  if      (which == 0) { X = query; W = Wq;  dst = qh;  }
  else if (which == 1) { X = key;   W = Wk1; dst = k1h; }
  else if (which == 2) { X = key;   W = Wk2; dst = k2h; }
  else                 { X = value; W = Wv;  dst = vh;  }
  const float* bvec = bias + which * EE;

  __shared__ _Float16 Xs[128 * 72];
  __shared__ _Float16 Ws[128 * 72];

  const int tid  = threadIdx.x;
  const int lane = tid & 31;
  const int wid  = tid >> 5;
  const int wy   = wid & 3;   // 4 wave-rows  -> 32 M rows each
  const int wx   = wid >> 2;  // 2 wave-cols  -> 64 N cols each
  const int m0   = blockIdx.y * 128;
  const int n0   = blockIdx.x * 128;

  v8f acc[2][4];
  #pragma unroll
  for (int a = 0; a < 2; ++a)
    #pragma unroll
    for (int c = 0; c < 4; ++c) acc[a][c] = (v8f){};

  for (int k0 = 0; k0 < EE; k0 += 64) {
    __syncthreads();
    for (int idx = tid; idx < 128 * 16; idx += 256) {
      const int row = idx >> 4, c4 = (idx & 15) << 2;
      const float4 xv = *(const float4*)(X + (size_t)(m0 + row) * EE + k0 + c4);
      v4h hx = { (_Float16)xv.x, (_Float16)xv.y, (_Float16)xv.z, (_Float16)xv.w };
      *(v4h*)&Xs[row * 72 + c4] = hx;
      const float4 wvec = *(const float4*)(W + (size_t)(n0 + row) * EE + k0 + c4);
      v4h hw = { (_Float16)wvec.x, (_Float16)wvec.y, (_Float16)wvec.z, (_Float16)wvec.w };
      *(v4h*)&Ws[row * 72 + c4] = hw;
      if (k0 + 64 < EE && (idx & 15) == 0) {   // prefetch next K tile rows into L2
        __builtin_prefetch(X + (size_t)(m0 + row) * EE + k0 + 64, 0, 1);
        __builtin_prefetch(W + (size_t)(n0 + row) * EE + k0 + 64, 0, 1);
      }
    }
    __syncthreads();
    #pragma unroll
    for (int kc = 0; kc < 64; kc += 32) {
      const v16h a0 = frag_a(&Xs[(wy * 32 +  0) * 72], 72, kc, lane);
      const v16h a1 = frag_a(&Xs[(wy * 32 + 16) * 72], 72, kc, lane);
      #pragma unroll
      for (int nt = 0; nt < 4; ++nt) {
        const v16h b = frag_b(&Ws[(wx * 64 + nt * 16) * 72], 72, kc, lane);
        acc[0][nt] = wmma_f16(a0, b, acc[0][nt]);
        acc[1][nt] = wmma_f16(a1, b, acc[1][nt]);
      }
    }
  }

  const int hf = lane >> 4, r = lane & 15;
  #pragma unroll
  for (int mt = 0; mt < 2; ++mt)
    #pragma unroll
    for (int nt = 0; nt < 4; ++nt)
      #pragma unroll
      for (int i = 0; i < 8; ++i) {
        const int M   = m0 + wy * 32 + mt * 16 + hf * 8 + i;
        const int col = n0 + wx * 64 + nt * 16 + r;
        const float val = acc[mt][nt][i] + bvec[col];
        const int t = M >> 2, b = M & 3, h = col >> 6, d = col & 63;
        dst[((size_t)((b * HH + h) * TT + t)) * HD + d] = (_Float16)val;
      }
}

// ---------------------------------------------------------------------------
// Kernel 2: row squared-norms of q/k1/k2 (arrays contiguous in workspace)
// grid: 3*RPA/256 = 768   block: 256, one thread per 64-element row
// ---------------------------------------------------------------------------
__global__ __launch_bounds__(256) void norms_kernel(
    const _Float16* __restrict__ qkk, float* __restrict__ norms)
{
  const int id = blockIdx.x * 256 + threadIdx.x;   // 0 .. 3*RPA-1
  const _Float16* row = qkk + (size_t)id * HD;
  float s = 0.f;
  #pragma unroll
  for (int c = 0; c < HD; c += 8) {
    const v8h x = *(const v8h*)&row[c];
    #pragma unroll
    for (int j = 0; j < 8; ++j) { const float f = (float)x[j]; s += f * f; }
  }
  norms[id] = s;
}

// ---------------------------------------------------------------------------
// Kernel 3: fused mixture-Gaussian-kernel attention, flash-style.
// Double-buffered: K1/K2 tiles staged with global_load_async_to_lds_b128
// (ASYNCcnt), V^T + norms pipelined through registers into the next buffer.
// Exponents are always <= 0 -> no running-max needed.
// grid: (T/128=8, B*H=64)   block: 256 (8 waves, 16 query rows each)
// ---------------------------------------------------------------------------
__global__ __launch_bounds__(256) void attn_kernel(
    const _Float16* __restrict__ qh,  const _Float16* __restrict__ k1h,
    const _Float16* __restrict__ k2h, const _Float16* __restrict__ vh,
    const float* __restrict__ norms,  const float* __restrict__ pi,
    _Float16* __restrict__ attn_out)
{
  __shared__ _Float16 Qs [128 * 72];
  __shared__ _Float16 K1s[2][128 * 72];
  __shared__ _Float16 K2s[2][128 * 72];
  __shared__ _Float16 Vt [2][64 * 136];     // v transposed: Vt[d][s]
  __shared__ _Float16 Es [8 * 16 * 136];    // per-wave e strip (16 x 128)
  __shared__ float    K1n[2][128];
  __shared__ float    K2n[2][128];

  const int tid  = threadIdx.x;
  const int lane = tid & 31;
  const int wid  = tid >> 5;
  const int hf   = lane >> 4, r = lane & 15;
  const int bh   = blockIdx.y;
  const int h    = bh & 15;
  const int b    = bh >> 4;
  const int q0   = blockIdx.x * 128;

  const float scaling = 0.125f;            // hd^-0.5
  const float c1 = 0.5f * scaling;         // scaling/2
  const float c2 = 1.5f * scaling;         // 3*scaling/2
  float p1 = fminf(fmaxf(fabsf(pi[h]),      1e-6f), 2.0f);
  float p2 = fminf(fmaxf(fabsf(pi[HH + h]), 1e-6f), 2.0f);

  const _Float16* k1base = k1h + (size_t)bh * TT * HD;
  const _Float16* k2base = k2h + (size_t)bh * TT * HD;
  const _Float16* vbase  = vh  + (size_t)bh * TT * HD;
  const float*    k1norm = norms + (size_t)RPA     + (size_t)bh * TT;
  const float*    k2norm = norms + (size_t)RPA * 2 + (size_t)bh * TT;

  // --- stage query block (128 x 64), synchronous one-time
  const _Float16* qbase = qh + ((size_t)bh * TT + q0) * HD;
  for (int idx = tid; idx < 128 * 8; idx += 256) {
    const int row = idx >> 3, c8 = (idx & 7) << 3;
    *(v8h*)&Qs[row * 72 + c8] = *(const v8h*)&qbase[(size_t)row * HD + c8];
  }

  // --- prologue: fill buffer 0 for s0 = 0
  #pragma unroll
  for (int j = 0; j < 4; ++j) {            // 128*8/256 = 4 chunks per thread
    const int idx = tid + j * 256;
    const int row = idx >> 3, c8 = (idx & 7) << 3;
    const unsigned go = (unsigned)((row * HD + c8) * sizeof(_Float16));
    async_copy_b128(lds_addr_of(&K1s[0][row * 72 + c8]), go, k1base);
    async_copy_b128(lds_addr_of(&K2s[0][row * 72 + c8]), go, k2base);
    const v8h vv = *(const v8h*)&vbase[row * HD + c8];
    #pragma unroll
    for (int e = 0; e < 8; ++e) Vt[0][(c8 + e) * 136 + row] = vv[e];
  }
  if (tid < 128) { K1n[0][tid] = k1norm[tid]; K2n[0][tid] = k2norm[tid]; }
  wait_async0();
  __syncthreads();

  const v16h aq0 = frag_a(&Qs[(wid * 16) * 72], 72,  0, lane);
  const v16h aq1 = frag_a(&Qs[(wid * 16) * 72], 72, 32, lane);

  float qn[8];
  const float* qnorm = norms + (size_t)bh * TT + q0;
  #pragma unroll
  for (int i = 0; i < 8; ++i) qn[i] = qnorm[wid * 16 + hf * 8 + i];

  v8f acc[4];
  #pragma unroll
  for (int dt = 0; dt < 4; ++dt) acc[dt] = (v8f){};
  float rs[8];
  #pragma unroll
  for (int i = 0; i < 8; ++i) rs[i] = 0.f;

  _Float16* Ew = &Es[wid * 16 * 136];
  const v8f zf = {};

  int cur = 0;
  for (int it = 0; it < TT / 128; ++it) {
    const int nxt = cur ^ 1;
    const int s1base = (it + 1) * 128;

    // ---- issue next tile: async K1/K2 copies + V/norm register prefetch
    v8h vreg[4];
    int vrow[4], vc8[4];
    float n1 = 0.f, n2 = 0.f;
    if (it + 1 < TT / 128) {
      const _Float16* k1b = k1base + (size_t)s1base * HD;
      const _Float16* k2b = k2base + (size_t)s1base * HD;
      const _Float16* vb  = vbase  + (size_t)s1base * HD;
      #pragma unroll
      for (int j = 0; j < 4; ++j) {
        const int idx = tid + j * 256;
        const int row = idx >> 3, c8 = (idx & 7) << 3;
        vrow[j] = row; vc8[j] = c8;
        const unsigned go = (unsigned)((row * HD + c8) * sizeof(_Float16));
        async_copy_b128(lds_addr_of(&K1s[nxt][row * 72 + c8]), go, k1b);
        async_copy_b128(lds_addr_of(&K2s[nxt][row * 72 + c8]), go, k2b);
        vreg[j] = *(const v8h*)&vb[row * HD + c8];
      }
      if (tid < 128) { n1 = k1norm[s1base + tid]; n2 = k2norm[s1base + tid]; }
    }

    // ---- S1 = q k1^T, S2 = q k2^T, e = pi1 e1 + pi2 e2 (current buffer)
    #pragma unroll
    for (int nt = 0; nt < 8; ++nt) {
      const v16h b1a = frag_b(&K1s[cur][(nt * 16) * 72], 72,  0, lane);
      const v16h b1b = frag_b(&K1s[cur][(nt * 16) * 72], 72, 32, lane);
      v8f s1 = wmma_f16(aq0, b1a, zf);
      s1     = wmma_f16(aq1, b1b, s1);
      const v16h b2a = frag_b(&K2s[cur][(nt * 16) * 72], 72,  0, lane);
      const v16h b2b = frag_b(&K2s[cur][(nt * 16) * 72], 72, 32, lane);
      v8f s2 = wmma_f16(aq0, b2a, zf);
      s2     = wmma_f16(aq1, b2b, s2);
      const float k1v = K1n[cur][nt * 16 + r];
      const float k2v = K2n[cur][nt * 16 + r];
      #pragma unroll
      for (int i = 0; i < 8; ++i) {
        const float d1 = qn[i] + k1v - 2.0f * s1[i];
        const float d2 = qn[i] + k2v - 2.0f * s2[i];
        const float e  = p1 * __expf(-c1 * d1) + p2 * __expf(-c2 * d2);
        rs[i] += e;
        Ew[(hf * 8 + i) * 136 + nt * 16 + r] = (_Float16)e;
      }
    }

    // Es is wave-private: a wave's DS ops execute in order through the LDS
    // pipe, so a wave-local dscnt wait suffices (no workgroup barrier).
    wait_ds0();

    // ---- PV: acc += e @ v (current buffer)
    #pragma unroll
    for (int kc = 0; kc < 128; kc += 32) {
      const v16h ae = frag_a(Ew, 136, kc, lane);
      #pragma unroll
      for (int dt = 0; dt < 4; ++dt) {
        const v16h bv = frag_b(&Vt[cur][(dt * 16) * 136], 136, kc, lane);
        acc[dt] = wmma_f16(ae, bv, acc[dt]);
      }
    }

    // ---- drain next-tile pipeline into buffer nxt, then workgroup barrier
    if (it + 1 < TT / 128) {
      #pragma unroll
      for (int j = 0; j < 4; ++j) {
        #pragma unroll
        for (int e = 0; e < 8; ++e)
          Vt[nxt][(vc8[j] + e) * 136 + vrow[j]] = vreg[j][e];
      }
      if (tid < 128) { K1n[nxt][tid] = n1; K2n[nxt][tid] = n2; }
      wait_async0();
    }
    __syncthreads();
    cur = nxt;
  }

  // reduce row sums across the 16 lanes that share each row (wave32 C layout)
  #pragma unroll
  for (int i = 0; i < 8; ++i) {
    float v = rs[i];
    v += __shfl_xor(v, 1, 32);
    v += __shfl_xor(v, 2, 32);
    v += __shfl_xor(v, 4, 32);
    v += __shfl_xor(v, 8, 32);
    rs[i] = v + 1e-6f;
  }

  #pragma unroll
  for (int dt = 0; dt < 4; ++dt)
    #pragma unroll
    for (int i = 0; i < 8; ++i) {
      const int trow = q0 + wid * 16 + hf * 8 + i;
      const int m    = trow * BB + b;
      const int col  = h * HD + dt * 16 + r;
      attn_out[(size_t)m * EE + col] = (_Float16)(acc[dt][i] / rs[i]);
    }
}

// ---------------------------------------------------------------------------
// Kernel 4: out = attn(f16) @ Wo^T + bo, f32 output
// grid: (8, 32)  block: 256
// ---------------------------------------------------------------------------
__global__ __launch_bounds__(256) void outproj_kernel(
    const _Float16* __restrict__ A, const float* __restrict__ W,
    const float* __restrict__ bias, float* __restrict__ out)
{
  __shared__ _Float16 As [128 * 72];
  __shared__ _Float16 Ws2[128 * 72];

  const int tid  = threadIdx.x;
  const int lane = tid & 31;
  const int wid  = tid >> 5;
  const int wy   = wid & 3;
  const int wx   = wid >> 2;
  const int m0   = blockIdx.y * 128;
  const int n0   = blockIdx.x * 128;

  v8f acc[2][4];
  #pragma unroll
  for (int a = 0; a < 2; ++a)
    #pragma unroll
    for (int c = 0; c < 4; ++c) acc[a][c] = (v8f){};

  for (int k0 = 0; k0 < EE; k0 += 64) {
    __syncthreads();
    for (int idx = tid; idx < 128 * 8; idx += 256) {
      const int row = idx >> 3, c8 = (idx & 7) << 3;
      *(v8h*)&As[row * 72 + c8] = *(const v8h*)&A[(size_t)(m0 + row) * EE + k0 + c8];
      if (k0 + 64 < EE && (idx & 7) == 0)
        __builtin_prefetch(A + (size_t)(m0 + row) * EE + k0 + 64, 0, 1);
    }
    for (int idx = tid; idx < 128 * 16; idx += 256) {
      const int row = idx >> 4, c4 = (idx & 15) << 2;
      const float4 wvec = *(const float4*)(W + (size_t)(n0 + row) * EE + k0 + c4);
      v4h hw = { (_Float16)wvec.x, (_Float16)wvec.y, (_Float16)wvec.z, (_Float16)wvec.w };
      *(v4h*)&Ws2[row * 72 + c4] = hw;
      if (k0 + 64 < EE && (idx & 15) == 0)
        __builtin_prefetch(W + (size_t)(n0 + row) * EE + k0 + 64, 0, 1);
    }
    __syncthreads();
    #pragma unroll
    for (int kc = 0; kc < 64; kc += 32) {
      const v16h a0 = frag_a(&As[(wy * 32 +  0) * 72], 72, kc, lane);
      const v16h a1 = frag_a(&As[(wy * 32 + 16) * 72], 72, kc, lane);
      #pragma unroll
      for (int nt = 0; nt < 4; ++nt) {
        const v16h bmat = frag_b(&Ws2[(wx * 64 + nt * 16) * 72], 72, kc, lane);
        acc[0][nt] = wmma_f16(a0, bmat, acc[0][nt]);
        acc[1][nt] = wmma_f16(a1, bmat, acc[1][nt]);
      }
    }
  }

  const int hf = lane >> 4, r = lane & 15;
  #pragma unroll
  for (int mt = 0; mt < 2; ++mt)
    #pragma unroll
    for (int nt = 0; nt < 4; ++nt)
      #pragma unroll
      for (int i = 0; i < 8; ++i) {
        const int M   = m0 + wy * 32 + mt * 16 + hf * 8 + i;
        const int col = n0 + wx * 64 + nt * 16 + r;
        out[(size_t)M * EE + col] = acc[mt][nt][i] + bias[col];
      }
}

// ---------------------------------------------------------------------------
extern "C" void kernel_launch(void* const* d_in, const int* in_sizes, int n_in,
                              void* d_out, int out_size, void* d_ws, size_t ws_size,
                              hipStream_t stream) {
  (void)in_sizes; (void)n_in; (void)out_size; (void)ws_size;
  const float* query = (const float*)d_in[0];
  const float* key   = (const float*)d_in[1];
  const float* value = (const float*)d_in[2];
  const float* Wq    = (const float*)d_in[3];
  const float* Wk1   = (const float*)d_in[4];
  const float* Wk2   = (const float*)d_in[5];
  const float* Wv    = (const float*)d_in[6];
  const float* bias  = (const float*)d_in[7];
  const float* Wo    = (const float*)d_in[8];
  const float* bo    = (const float*)d_in[9];
  const float* pi    = (const float*)d_in[10];

  char* ws = (char*)d_ws;
  const size_t P = (size_t)RPA * HD * sizeof(_Float16);   // 8 MiB per head-major tensor
  _Float16* qh   = (_Float16*)(ws + 0 * P);
  _Float16* k1h  = (_Float16*)(ws + 1 * P);
  _Float16* k2h  = (_Float16*)(ws + 2 * P);
  _Float16* vh   = (_Float16*)(ws + 3 * P);
  float*    nrms = (float*)   (ws + 4 * P);               // 3*RPA f32
  _Float16* attn = (_Float16*)(ws + 4 * P + (size_t)3 * RPA * sizeof(float));

  proj_kernel<<<dim3(8, 32, 4), 256, 0, stream>>>(query, key, value,
      Wq, Wk1, Wk2, Wv, bias, qh, k1h, k2h, vh);
  norms_kernel<<<dim3(3 * RPA / 256), 256, 0, stream>>>(qh, nrms);
  attn_kernel<<<dim3(8, 64), 256, 0, stream>>>(qh, k1h, k2h, vh, nrms, pi, attn);
  outproj_kernel<<<dim3(8, 32), 256, 0, stream>>>(attn, Wo, bo, (float*)d_out);
}
